// MMDLoss_11106785428164
// MI455X (gfx1250) — compile-verified, matrix operand order checked by reading, not run on hardware
//
#include <hip/hip_runtime.h>
#include <hip/hip_bf16.h>

typedef __bf16 bf16_t;
typedef bf16_t v16bf __attribute__((ext_vector_type(16)));
typedef float  v8f   __attribute__((ext_vector_type(8)));
typedef unsigned int v4u __attribute__((ext_vector_type(4)));

#define GAMMA 0.05f
#define DIM 128

// ---------------------------------------------------------------------------
__global__ void init_accum_kernel(float* acc) {
    if (threadIdx.x < 128) acc[threadIdx.x] = 0.0f;
}

// ---------------------------------------------------------------------------
// Round f32 rows to bf16 and compute row squared-norms FROM THE ROUNDED data
// (diagonal d2 then cancels to ~0, matching the reference's max(d2,0) clamp).
__global__ void prep_rows_kernel(const float* __restrict__ src,
                                 bf16_t* __restrict__ dst,
                                 float* __restrict__ norms, int nrows) {
    int row = blockIdx.x * blockDim.x + threadIdx.x;
    if (row >= nrows) return;
    const float* s = src + (size_t)row * DIM;
    bf16_t* o = dst + (size_t)row * DIM;
    float acc = 0.0f;
#pragma unroll 4
    for (int k = 0; k < DIM; k += 4) {
        float4 v = *(const float4*)(s + k);
        bf16_t b0 = (bf16_t)v.x, b1 = (bf16_t)v.y, b2 = (bf16_t)v.z, b3 = (bf16_t)v.w;
        float f0 = (float)b0, f1 = (float)b1, f2 = (float)b2, f3 = (float)b3;
        acc += f0 * f0 + f1 * f1 + f2 * f2 + f3 * f3;
        o[k] = b0; o[k + 1] = b1; o[k + 2] = b2; o[k + 3] = b3;
    }
    norms[row] = acc;
}

// ---------------------------------------------------------------------------
__global__ void wsum_kernel(const float* __restrict__ w, int n, float* out) {
    int i = blockIdx.x * blockDim.x + threadIdx.x;
    float v = (i < n) ? w[i] : 0.0f;
    for (int o = 16; o; o >>= 1) v += __shfl_down(v, o, 32);
    if ((threadIdx.x & 31) == 0) atomicAdd(out, v);
}

// ---------------------------------------------------------------------------
// Per-16x16-tile epilogue: sum_{elems} f_sym * wa[i] * exp(-g*max(aa+bb-2dot,0))
// (wb factor applied by the caller, it is constant per lane per tile column).
template <bool SYM, bool HAS_WA>
__device__ __forceinline__ float tile_sum(v8f c, int ti, int tj, int r, int half,
                                          const float* __restrict__ aa,
                                          const float* __restrict__ wa, float bbj) {
    if (SYM && tj < ti) return 0.0f;   // tile strictly below diagonal
    float s = 0.0f;
#pragma unroll
    for (int v = 0; v < 8; ++v) {
        const int il = half * 8 + v;       // C/D layout: lanes16-31 hold M=v+8
        const int i = ti * 16 + il;
        float d2 = fmaxf(aa[i] + bbj - 2.0f * c[v], 0.0f);
        float kv = __expf(-GAMMA * d2);
        float f = 1.0f;
        if (SYM) f = (ti == tj) ? ((r > il) ? 2.0f : ((r == il) ? 1.0f : 0.0f))
                                : 2.0f;
        float w = HAS_WA ? wa[i] : 1.0f;
        s += f * w * kv;
    }
    return s;
}

// ---------------------------------------------------------------------------
// Fused WMMA kernel with 2x2 register blocking per wave:
//   block = 256 threads = 8 waves arranged 2x4; block tile = 64 x 128.
//   Each wave computes a 32x32 output region = 2x2 WMMA tiles -> every A/B
//   fragment feeds two v_wmma_f32_16x16x32_bf16, 16 WMMAs per K-sweep.
template <bool SYM, bool HAS_WA, bool HAS_WB>
__global__ __launch_bounds__(256) void rbf_tile_kernel(
    const bf16_t* __restrict__ A, const bf16_t* __restrict__ B,
    const float* __restrict__ aa, const float* __restrict__ bb,
    const float* __restrict__ wa, const float* __restrict__ wb,
    float* __restrict__ accum) {
    const int wave = threadIdx.x >> 5;
    const int lane = threadIdx.x & 31;
    const int tiBase = blockIdx.x * 4;   // tile-row base (4 tile rows / block)
    const int tjBase = blockIdx.y * 8;   // tile-col base (8 tile cols / block)

    if (SYM && (tjBase + 7) < tiBase) return;   // whole block below diagonal

    const int ti0 = tiBase + (wave >> 2) * 2;   // wave tiles: ti0, ti0+1
    const int tj0 = tjBase + (wave & 3) * 2;    //             tj0, tj0+1
    const int r = lane & 15;                    // M (A) / N (B) index in tile
    const int half = lane >> 4;                 // K-split half per ISA layout

    float acc = 0.0f;
    if (!(SYM && (tj0 + 1) < ti0)) {
        const bf16_t* a0p = A + ((size_t)(ti0 * 16 + r)) * DIM + half * 8;
        const bf16_t* a1p = a0p + (size_t)16 * DIM;
        const bf16_t* b0p = B + ((size_t)(tj0 * 16 + r)) * DIM + half * 8;
        const bf16_t* b1p = b0p + (size_t)16 * DIM;

        v8f c00 = {0.f,0.f,0.f,0.f,0.f,0.f,0.f,0.f};
        v8f c01 = c00, c10 = c00, c11 = c00;
#pragma unroll
        for (int k0 = 0; k0 < DIM; k0 += 32) {
            union { v4u q[2]; v16bf v; } a0, a1, b0, b1;
            a0.q[0] = *(const v4u*)(a0p + k0);       // K = k0+half*8   .. +7
            a0.q[1] = *(const v4u*)(a0p + k0 + 16);  // K = k0+half*8+16.. +7
            a1.q[0] = *(const v4u*)(a1p + k0);
            a1.q[1] = *(const v4u*)(a1p + k0 + 16);
            b0.q[0] = *(const v4u*)(b0p + k0);
            b0.q[1] = *(const v4u*)(b0p + k0 + 16);
            b1.q[0] = *(const v4u*)(b1p + k0);
            b1.q[1] = *(const v4u*)(b1p + k0 + 16);
            c00 = __builtin_amdgcn_wmma_f32_16x16x32_bf16(false, a0.v, false, b0.v, (short)0, c00, false, false);
            c01 = __builtin_amdgcn_wmma_f32_16x16x32_bf16(false, a0.v, false, b1.v, (short)0, c01, false, false);
            c10 = __builtin_amdgcn_wmma_f32_16x16x32_bf16(false, a1.v, false, b0.v, (short)0, c10, false, false);
            c11 = __builtin_amdgcn_wmma_f32_16x16x32_bf16(false, a1.v, false, b1.v, (short)0, c11, false, false);
        }

        const int j0 = tj0 * 16 + r, j1 = j0 + 16;
        const float bb0 = bb[j0], bb1 = bb[j1];
        const float wb0 = HAS_WB ? wb[j0] : 1.0f;
        const float wb1 = HAS_WB ? wb[j1] : 1.0f;
        acc += wb0 * tile_sum<SYM, HAS_WA>(c00, ti0,     tj0,     r, half, aa, wa, bb0);
        acc += wb1 * tile_sum<SYM, HAS_WA>(c01, ti0,     tj0 + 1, r, half, aa, wa, bb1);
        acc += wb0 * tile_sum<SYM, HAS_WA>(c10, ti0 + 1, tj0,     r, half, aa, wa, bb0);
        acc += wb1 * tile_sum<SYM, HAS_WA>(c11, ti0 + 1, tj0 + 1, r, half, aa, wa, bb1);
    }

    // wave -> LDS -> block reduction, one atomic per block (32-slot scatter)
    for (int o = 16; o; o >>= 1) acc += __shfl_down(acc, o, 32);
    __shared__ float lds[8];
    if (lane == 0) lds[wave] = acc;
    __syncthreads();
    if (threadIdx.x == 0) {
        float s = 0.0f;
#pragma unroll
        for (int wv = 0; wv < 8; ++wv) s += lds[wv];
        atomicAdd(&accum[blockIdx.x & 31], s);
    }
}

// ---------------------------------------------------------------------------
// mmd = xx/S^2 + yy/m^2 - 2*xy/(S*m); out = sqrt(mmd)
__global__ void finalize_kernel(const float* __restrict__ acc, float* out, float mcount) {
    float xx = 0.f, yy = 0.f, xy = 0.f;
#pragma unroll
    for (int i = 0; i < 32; ++i) { xx += acc[i]; yy += acc[32 + i]; xy += acc[64 + i]; }
    float S = acc[96];
    float mmd = xx / (S * S) + yy / (mcount * mcount) - 2.0f * xy / (S * mcount);
    *out = sqrtf(fmaxf(mmd, 0.0f));
}

// ---------------------------------------------------------------------------
extern "C" void kernel_launch(void* const* d_in, const int* in_sizes, int n_in,
                              void* d_out, int out_size, void* d_ws, size_t ws_size,
                              hipStream_t stream) {
    const float* Nf = (const float*)d_in[0];
    const float* Rf = (const float*)d_in[1];
    const float* W  = (const float*)d_in[2];
    const int n = in_sizes[0] / DIM;   // 8192
    const int m = in_sizes[1] / DIM;   // 8192

    // workspace layout
    float*  accum = (float*)d_ws;                 // xx[32], yy[32], xy[32], S
    bf16_t* Nbf   = (bf16_t*)((char*)d_ws + 512);
    bf16_t* Rbf   = Nbf + (size_t)n * DIM;
    float*  aa    = (float*)(Rbf + (size_t)m * DIM);
    float*  bb    = aa + n;

    init_accum_kernel<<<1, 128, 0, stream>>>(accum);
    prep_rows_kernel<<<(n + 255) / 256, 256, 0, stream>>>(Nf, Nbf, aa, n);
    prep_rows_kernel<<<(m + 255) / 256, 256, 0, stream>>>(Rf, Rbf, bb, m);
    wsum_kernel<<<(n + 255) / 256, 256, 0, stream>>>(W, n, accum + 96);

    // xx: N x N, symmetric, weights on both sides (unnormalized)
    rbf_tile_kernel<true, true, true><<<dim3(n / 64, n / 128), 256, 0, stream>>>(
        Nbf, Nbf, aa, aa, W, W, accum + 0);
    // yy: R x R, symmetric, uniform weights
    rbf_tile_kernel<true, false, false><<<dim3(m / 64, m / 128), 256, 0, stream>>>(
        Rbf, Rbf, bb, bb, nullptr, nullptr, accum + 32);
    // xy: N x R, full, weights on rows only
    rbf_tile_kernel<false, true, false><<<dim3(n / 64, m / 128), 256, 0, stream>>>(
        Nbf, Rbf, aa, bb, W, nullptr, accum + 64);

    finalize_kernel<<<1, 1, 0, stream>>>(accum, (float*)d_out, (float)m);
}
